// MultiHeadAttention_67095979098528
// MI455X (gfx1250) — matrix operand order repaired and checked by reading, not compiled, not run on hardware
//
#include <hip/hip_runtime.h>
#include <hip/hip_bf16.h>

typedef __bf16 bf16;
typedef __attribute__((ext_vector_type(16))) __bf16 v16bf;
typedef __attribute__((ext_vector_type(8)))  float  v8f;

#define D_MODEL 1024
#define NHEAD   16
#define DK      64
#define BATCH   2
#define SLEN    2048
#define MROWS   (BATCH * SLEN)   // 4096

// ---------------- DPP16 butterfly reductions over the 16-lane half ----------------
// Steps: lane^1, lane^2 (quad_perm), lane^7-in-8 (row_half_mirror), lane^15-in-16
// (row_mirror). All lanes in a 16-lane DPP row end with the full reduction.

__device__ inline float dpp_reduce_max16(float x) {
    int t;
    t = __builtin_amdgcn_update_dpp(0, __float_as_int(x), 0xB1, 0xF, 0xF, true);  // quad_perm [1,0,3,2]
    x = fmaxf(x, __int_as_float(t));
    t = __builtin_amdgcn_update_dpp(0, __float_as_int(x), 0x4E, 0xF, 0xF, true);  // quad_perm [2,3,0,1]
    x = fmaxf(x, __int_as_float(t));
    t = __builtin_amdgcn_update_dpp(0, __float_as_int(x), 0x141, 0xF, 0xF, true); // row_half_mirror
    x = fmaxf(x, __int_as_float(t));
    t = __builtin_amdgcn_update_dpp(0, __float_as_int(x), 0x140, 0xF, 0xF, true); // row_mirror
    x = fmaxf(x, __int_as_float(t));
    return x;
}

__device__ inline float dpp_reduce_add16(float x) {
    int t;
    t = __builtin_amdgcn_update_dpp(0, __float_as_int(x), 0xB1, 0xF, 0xF, true);
    x = x + __int_as_float(t);
    t = __builtin_amdgcn_update_dpp(0, __float_as_int(x), 0x4E, 0xF, 0xF, true);
    x = x + __int_as_float(t);
    t = __builtin_amdgcn_update_dpp(0, __float_as_int(x), 0x141, 0xF, 0xF, true);
    x = x + __int_as_float(t);
    t = __builtin_amdgcn_update_dpp(0, __float_as_int(x), 0x140, 0xF, 0xF, true);
    x = x + __int_as_float(t);
    return x;
}

// ---------------- WMMA fragment loaders (gfx1250 wave32 layouts) ----------------

// A-matrix 16x32 (MxK) bf16, row-major source with row stride ld.
__device__ inline v16bf load_a_frag(const bf16* __restrict__ src, int ld) {
    int lane = threadIdx.x & 31;
    int m = lane & 15, half = lane >> 4;
    v16bf a;
#pragma unroll
    for (int j = 0; j < 8; ++j) {
        int k = (half << 4) + (j << 1);
        a[2 * j]     = src[m * ld + k];
        a[2 * j + 1] = src[m * ld + k + 1];
    }
    return a;
}

// B-matrix 32x16 (KxN) bf16, row-major source with row stride ld.
__device__ inline v16bf load_b_frag(const bf16* __restrict__ src, int ld) {
    int lane = threadIdx.x & 31;
    int n = lane & 15, half = lane >> 4;
    v16bf b;
#pragma unroll
    for (int j = 0; j < 8; ++j) {
        int k = (half << 4) + (j << 1);
        b[2 * j]     = src[k * ld + n];
        b[2 * j + 1] = src[(k + 1) * ld + n];
    }
    return b;
}

// B-matrix 32x16 = transpose of a row-major 16x32 tile: B[k][n] = src[n*ld + k]
__device__ inline v16bf load_bt_frag(const bf16* __restrict__ src, int ld) {
    int lane = threadIdx.x & 31;
    int n = lane & 15, half = lane >> 4;
    v16bf b;
#pragma unroll
    for (int j = 0; j < 8; ++j) {
        int k = (half << 4) + (j << 1);
        b[2 * j]     = src[n * ld + k];
        b[2 * j + 1] = src[n * ld + k + 1];
    }
    return b;
}

// ---------------- conversion kernels ----------------

__global__ void f32_to_bf16_kernel(const float* __restrict__ in, bf16* __restrict__ out, int n) {
    int i = blockIdx.x * blockDim.x + threadIdx.x;
    if (i < n) out[i] = (bf16)in[i];
}

// [H, D, DK] fp32 -> [D, H*DK] bf16 (fold per-head projection into one matrix)
__global__ void fold_w_kernel(const float* __restrict__ in, bf16* __restrict__ out) {
    int i = blockIdx.x * blockDim.x + threadIdx.x;   // over D_MODEL*D_MODEL
    int d = i >> 10;
    int c = i & 1023;
    int h = c >> 6;
    int kk = c & 63;
    out[i] = (bf16)in[(h << 16) + (d << 6) + kk];    // h*D*DK + d*DK + kk
}

// ---------------- WMMA GEMM: C[M,N] = A[M,K](bf16) x B[K,N](bf16) ----------------

template <bool OUT_BF16>
__global__ void gemm_bf16_kernel(const bf16* __restrict__ A, const bf16* __restrict__ B,
                                 void* __restrict__ Cp, int M, int N, int K) {
    int wid = blockIdx.x * (blockDim.x >> 5) + (threadIdx.x >> 5);
    int ntiles = N >> 6;
    int mt = wid / ntiles, nt = wid % ntiles;
    int m0 = mt << 4, n0 = nt << 6;
    if (m0 >= M) return;

    v8f c[4] = {v8f{}, v8f{}, v8f{}, v8f{}};
    for (int k0 = 0; k0 < K; k0 += 32) {
        v16bf a = load_a_frag(A + (size_t)m0 * K + k0, K);
#pragma unroll
        for (int i = 0; i < 4; ++i) {
            v16bf b = load_b_frag(B + (size_t)k0 * N + n0 + i * 16, N);
            c[i] = __builtin_amdgcn_wmma_f32_16x16x32_bf16(false, a, false, b,
                                                           (short)0, c[i], false, false);
        }
    }

    int lane = threadIdx.x & 31;
    int col = lane & 15, half = lane >> 4;
#pragma unroll
    for (int i = 0; i < 4; ++i) {
#pragma unroll
        for (int j = 0; j < 8; ++j) {
            int row = (half << 3) + j;
            size_t idx = (size_t)(m0 + row) * N + n0 + i * 16 + col;
            if (OUT_BF16) ((bf16*)Cp)[idx] = (bf16)c[i][j];
            else          ((float*)Cp)[idx] = c[i][j];
        }
    }
}

// ---------------- flash attention: one wave = 16 query rows of one (b,h) ----------------

__global__ void attn_kernel(const bf16* __restrict__ qb, const bf16* __restrict__ kb,
                            const bf16* __restrict__ vb, const float* __restrict__ mask,
                            bf16* __restrict__ ob) {
    __shared__ bf16 lds_p[4][16][36];   // per-wave 16x32 P tile (padded rows)

    int warp = threadIdx.x >> 5;
    int wid  = blockIdx.x * 4 + warp;
    int qt = wid & 127;
    int h  = (wid >> 7) & 15;
    int b  = wid >> 11;
    int q0 = qt << 4;

    const int ld = D_MODEL;
    const bf16* qh = qb + (size_t)b * SLEN * ld + h * DK;
    const bf16* kh = kb + (size_t)b * SLEN * ld + h * DK;
    const bf16* vh = vb + (size_t)b * SLEN * ld + h * DK;

    int lane = threadIdx.x & 31;
    int col = lane & 15, half = lane >> 4;

    v16bf a_lo = load_a_frag(qh + (size_t)q0 * ld, ld);
    v16bf a_hi = load_a_frag(qh + (size_t)q0 * ld + 32, ld);

    v8f acc[4] = {v8f{}, v8f{}, v8f{}, v8f{}};
    float mrow[8], lsum[8];   // lsum is a per-lane PARTIAL sum (reduced once at end)
#pragma unroll
    for (int j = 0; j < 8; ++j) { mrow[j] = -1e30f; lsum[j] = 0.0f; }

    const float scale = 0.125f;  // 1/sqrt(64)
    const float* mrow_base = mask + (size_t)(q0 + (half << 3)) * SLEN + col;

    for (int s0 = 0; s0 < SLEN; s0 += 32) {
        // preload mask values for this chunk (lets the compiler clause the loads)
        float mk0[8], mk1[8];
#pragma unroll
        for (int j = 0; j < 8; ++j) {
            mk0[j] = mrow_base[(size_t)j * SLEN + s0];
            mk1[j] = mrow_base[(size_t)j * SLEN + s0 + 16];
        }

        // scores: 16 q-rows x 32 keys, K-dim = dk = 64
        v8f sc0 = {}, sc1 = {};
        {
            v16bf b0  = load_bt_frag(kh + (size_t)s0 * ld, ld);
            v16bf b0h = load_bt_frag(kh + (size_t)s0 * ld + 32, ld);
            sc0 = __builtin_amdgcn_wmma_f32_16x16x32_bf16(false, a_lo, false, b0,  (short)0, sc0, false, false);
            sc0 = __builtin_amdgcn_wmma_f32_16x16x32_bf16(false, a_hi, false, b0h, (short)0, sc0, false, false);
            v16bf b1  = load_bt_frag(kh + (size_t)(s0 + 16) * ld, ld);
            v16bf b1h = load_bt_frag(kh + (size_t)(s0 + 16) * ld + 32, ld);
            sc1 = __builtin_amdgcn_wmma_f32_16x16x32_bf16(false, a_lo, false, b1,  (short)0, sc1, false, false);
            sc1 = __builtin_amdgcn_wmma_f32_16x16x32_bf16(false, a_hi, false, b1h, (short)0, sc1, false, false);
        }

        float v0[8], v1[8], pmax[8];
#pragma unroll
        for (int j = 0; j < 8; ++j) {
            v0[j] = sc0[j] * scale + mk0[j];
            v1[j] = sc1[j] * scale + mk1[j];
            pmax[j] = fmaxf(v0[j], v1[j]);
        }

        // online softmax: only the row-max needs a cross-lane reduction per chunk
#pragma unroll
        for (int j = 0; j < 8; ++j) {
            int row = (half << 3) + j;

            float mx    = dpp_reduce_max16(pmax[j]);
            float mnew  = fmaxf(mrow[j], mx);
            float alpha = __expf(mrow[j] - mnew);
            mrow[j] = mnew;

            float e0 = __expf(v0[j] - mnew);
            float e1 = __expf(v1[j] - mnew);

            lsum[j] = lsum[j] * alpha + e0 + e1;   // per-lane partial; alpha lane-uniform
#pragma unroll
            for (int i = 0; i < 4; ++i)
                acc[i][j] = acc[i][j] * alpha;

            lds_p[warp][row][col]      = (bf16)e0;
            lds_p[warp][row][col + 16] = (bf16)e1;
        }

        asm volatile("s_wait_dscnt 0" ::: "memory");

        // P(16x32) @ V(32x64) -> accumulate
        v16bf pa = load_a_frag(&lds_p[warp][0][0], 36);
#pragma unroll
        for (int i = 0; i < 4; ++i) {
            v16bf bv = load_b_frag(vh + (size_t)s0 * ld + i * 16, ld);
            acc[i] = __builtin_amdgcn_wmma_f32_16x16x32_bf16(false, pa, false, bv,
                                                             (short)0, acc[i], false, false);
        }
    }

    // final denominator: reduce per-lane partial sums across the 16-lane half
    float lrow[8];
#pragma unroll
    for (int j = 0; j < 8; ++j)
        lrow[j] = dpp_reduce_add16(lsum[j]);

    // normalize and store O (bf16, [B*S, H*DK] layout for the output GEMM)
#pragma unroll
    for (int i = 0; i < 4; ++i) {
#pragma unroll
        for (int j = 0; j < 8; ++j) {
            int row = (half << 3) + j;
            float val = acc[i][j] / lrow[j];
            ob[((size_t)b * SLEN + q0 + row) * D_MODEL + h * DK + i * 16 + col] = (bf16)val;
        }
    }
}

// ---------------- host launch ----------------

extern "C" void kernel_launch(void* const* d_in, const int* in_sizes, int n_in,
                              void* d_out, int out_size, void* d_ws, size_t ws_size,
                              hipStream_t stream) {
    (void)in_sizes; (void)n_in; (void)out_size; (void)ws_size;

    const float* Q    = (const float*)d_in[0];
    const float* K    = (const float*)d_in[1];
    const float* V    = (const float*)d_in[2];
    const float* mask = (const float*)d_in[3];
    // d_in[4] key_padding_mask: ignored by the reference module
    const float* WQ   = (const float*)d_in[5];
    const float* WK   = (const float*)d_in[6];
    const float* WV   = (const float*)d_in[7];
    const float* WO   = (const float*)d_in[8];
    float* out = (float*)d_out;

    const size_t XN = (size_t)MROWS * D_MODEL;
    const size_t WN = (size_t)D_MODEL * D_MODEL;

    char* ws = (char*)d_ws;
    bf16* Xq = (bf16*)ws; ws += XN * 2;
    bf16* Xk = (bf16*)ws; ws += XN * 2;
    bf16* Xv = (bf16*)ws; ws += XN * 2;
    bf16* Wq = (bf16*)ws; ws += WN * 2;
    bf16* Wk = (bf16*)ws; ws += WN * 2;
    bf16* Wv = (bf16*)ws; ws += WN * 2;
    bf16* Wo = (bf16*)ws; ws += WN * 2;
    bf16* qb = (bf16*)ws; ws += XN * 2;
    bf16* kb = (bf16*)ws; ws += XN * 2;
    bf16* vb = (bf16*)ws; ws += XN * 2;
    bf16* ob = (bf16*)ws; ws += XN * 2;

    // 1) fp32 -> bf16 conversions
    f32_to_bf16_kernel<<<(int)(XN / 256), 256, 0, stream>>>(Q, Xq, (int)XN);
    f32_to_bf16_kernel<<<(int)(XN / 256), 256, 0, stream>>>(K, Xk, (int)XN);
    f32_to_bf16_kernel<<<(int)(XN / 256), 256, 0, stream>>>(V, Xv, (int)XN);
    fold_w_kernel<<<(int)(WN / 256), 256, 0, stream>>>(WQ, Wq);
    fold_w_kernel<<<(int)(WN / 256), 256, 0, stream>>>(WK, Wk);
    fold_w_kernel<<<(int)(WN / 256), 256, 0, stream>>>(WV, Wv);
    f32_to_bf16_kernel<<<(int)(WN / 256), 256, 0, stream>>>(WO, Wo, (int)WN);

    // 2) QKV projections: [4096,1024] x [1024,1024] (bf16 out)
    gemm_bf16_kernel<true><<<1024, 128, 0, stream>>>(Xq, Wq, qb, MROWS, D_MODEL, D_MODEL);
    gemm_bf16_kernel<true><<<1024, 128, 0, stream>>>(Xk, Wk, kb, MROWS, D_MODEL, D_MODEL);
    gemm_bf16_kernel<true><<<1024, 128, 0, stream>>>(Xv, Wv, vb, MROWS, D_MODEL, D_MODEL);

    // 3) flash attention
    attn_kernel<<<1024, 128, 0, stream>>>(qb, kb, vb, mask, ob);

    // 4) output projection (fp32 out -> d_out)
    gemm_bf16_kernel<false><<<1024, 128, 0, stream>>>(ob, Wo, out, MROWS, D_MODEL, D_MODEL);
}
